// BertAttention_56556129354021
// MI455X (gfx1250) — matrix operand order repaired and checked by reading, not compiled, not run on hardware
//
#include <hip/hip_runtime.h>

// ---------------------------------------------------------------------------
// BERT attention block for MI455X (gfx1250, wave32, WMMA + async LDS copies).
// - fp32 operands converted once to fp16; all GEMMs run v_wmma_f32_16x16x32_f16
//   (f16 in, f32 accumulate).
// - GEMM tiles staged with GLOBAL_LOAD_ASYNC_TO_LDS_B128 (ASYNCcnt), double
//   buffered so the copy of tile i+1 overlaps the WMMAs of tile i.
// - Attention is flash-style online softmax (scores never touch HBM).
// Workspace (64 MiB): Xh(8M) Wq/Wk/Wv/Wo(2M each) Qh/Kh/Vh/Ctx(8M each) Y(16M).
// ---------------------------------------------------------------------------

typedef _Float16 half8  __attribute__((ext_vector_type(8)));
typedef _Float16 half16 __attribute__((ext_vector_type(16)));
typedef float    f32x8  __attribute__((ext_vector_type(8)));
typedef float    f32x4  __attribute__((ext_vector_type(4)));

union Frag {
    half16   v;
    half8    h8[2];
    _Float16 e[16];
};

__device__ __forceinline__ f32x8 wmma16x16x32(half16 a, half16 b, f32x8 c) {
    // D = A(16x32 f16) * B(32x16 f16) + C(16x16 f32)
    return __builtin_amdgcn_wmma_f32_16x16x32_f16(
        /*neg_a=*/false, a, /*neg_b=*/false, b,
        /*c_mod=*/(short)0, c, /*reuse_a=*/false, /*reuse_b=*/false);
}

// Async 16-byte global -> LDS copy (ASYNCcnt tracked). GV addressing mode.
__device__ __forceinline__ void async_copy_b128(void* lds_ptr, const void* gptr) {
    const unsigned int lds = (unsigned int)(unsigned long long)lds_ptr;
    const unsigned long long ga = (unsigned long long)gptr;
    asm volatile("global_load_async_to_lds_b128 %0, %1, off"
                 :: "v"(lds), "v"(ga) : "memory");
}
__device__ __forceinline__ void wait_async_zero() {
    asm volatile("s_wait_asynccnt 0" ::: "memory");
}

// ---------------------------------------------------------------------------
// fp32 -> fp16 bulk convert (n multiple of 8). One-time prolog for X and W.
// ---------------------------------------------------------------------------
__global__ __launch_bounds__(256) void cvt_f32_f16_kernel(
    const float* __restrict__ src, _Float16* __restrict__ dst, int n)
{
    const int i = (blockIdx.x * 256 + threadIdx.x) * 8;
    if (i < n) {
        f32x4 a = *(const f32x4*)(src + i);
        f32x4 b = *(const f32x4*)(src + i + 4);
        Frag t;
        #pragma unroll
        for (int j = 0; j < 4; ++j) {
            t.e[j]     = (_Float16)a[j];
            t.e[4 + j] = (_Float16)b[j];
        }
        *(half8*)(dst + i) = t.h8[0];
    }
}

// ---------------------------------------------------------------------------
// Tiled WMMA GEMM, all-f16 operands:  out = X[M,K] * W[N,K]^T + bias (+res)
// Block: 256 threads (8 waves), tile 128(M) x 64(N), K-step 32.
// Tiles staged via async global->LDS b128 copies, double buffered.
// Wave (4 along M, 2 along N) owns a 32x32 subtile = 2x2 WMMA accumulators.
// ---------------------------------------------------------------------------
constexpr int GM = 128, GN = 64, GK = 32;
constexpr int ASTR = 40;   // LDS row stride in halves (80B: bank-skew, 16B-aligned)
constexpr int BSTRD = 40;

__global__ __launch_bounds__(256) void wmma_gemm_async_kernel(
    const _Float16* __restrict__ X, const _Float16* __restrict__ W,
    const float* __restrict__ bias, const float* __restrict__ residual,
    _Float16* __restrict__ outH, float* __restrict__ outF,
    int M, int N, int K)
{
    __shared__ __attribute__((aligned(16))) _Float16 Ah[2][GM * ASTR];
    __shared__ __attribute__((aligned(16))) _Float16 Bh[2][GN * BSTRD];

    const int tid  = threadIdx.x;
    const int wave = tid >> 5, lane = tid & 31;
    const int hf   = lane >> 4, l16 = lane & 15;
    const int waveM = wave >> 1, waveN = wave & 1;
    const int m0 = blockIdx.y * GM, n0 = blockIdx.x * GN;

    // async-copy assignments (16B chunks):
    // A tile: 128 rows x 64B -> thread t copies 32B of row t>>1
    const int arow = tid >> 1, acb = (tid & 1) * 16;   // halves offset in row
    // B tile: 64 rows x 64B -> thread t copies 16B of row t>>2
    const int brow = tid >> 2, bcb = (tid & 3) * 8;

    const _Float16* xrow = X + (size_t)(m0 + arow) * K + acb;
    const _Float16* wrow = W + (size_t)(n0 + brow) * K + bcb;

    auto issue_tile = [&](int buf, int k0) {
        void* la = (void*)&Ah[buf][arow * ASTR + acb];
        async_copy_b128(la, xrow + k0);
        async_copy_b128((void*)((char*)la + 16), xrow + k0 + 8);
        async_copy_b128((void*)&Bh[buf][brow * BSTRD + bcb], wrow + k0);
    };

    f32x8 acc[2][2] = {};
    const int nk = K / GK;
    issue_tile(0, 0);  // prologue copy

    for (int i = 0; i < nk; ++i) {
        wait_async_zero();   // this wave's copies into buf[i&1] landed
        __syncthreads();     // all waves' copies landed; prev reads of nxt done
        if (i + 1 < nk) issue_tile((i + 1) & 1, (i + 1) * GK);  // overlap w/ WMMA
        const int buf = i & 1;

        // per-lane fragment loads (ISA 16-bit A/B layouts)
        Frag af[2], bf[2];
        #pragma unroll
        for (int mt = 0; mt < 2; ++mt) {
            const _Float16* ap = &Ah[buf][(waveM * 32 + mt * 16 + l16) * ASTR];
            af[mt].h8[0] = *(const half8*)(ap + hf * 8);        // K {0-7}/{8-15}
            af[mt].h8[1] = *(const half8*)(ap + 16 + hf * 8);   // K {16-23}/{24-31}
        }
        #pragma unroll
        for (int nt = 0; nt < 2; ++nt) {
            const _Float16* bp =
                &Bh[buf][(waveN * 32 + nt * 16 + l16) * BSTRD + hf * 8];
            bf[nt].h8[0] = *(const half8*)(bp);                 // K {0-7}/{8-15}
            bf[nt].h8[1] = *(const half8*)(bp + 16);            // K {16-23}/{24-31}
        }
        #pragma unroll
        for (int mt = 0; mt < 2; ++mt)
            #pragma unroll
            for (int nt = 0; nt < 2; ++nt)
                acc[mt][nt] = wmma16x16x32(af[mt].v, bf[nt].v, acc[mt][nt]);
    }

    // epilogue: C layout VGPR r -> row r + 8*hf, col l16
    #pragma unroll
    for (int nt = 0; nt < 2; ++nt) {
        const int col = n0 + waveN * 32 + nt * 16 + l16;
        const float bv = bias[col];
        #pragma unroll
        for (int mt = 0; mt < 2; ++mt) {
            #pragma unroll
            for (int r = 0; r < 8; ++r) {
                const int row = m0 + waveM * 32 + mt * 16 + r + hf * 8;
                const size_t o = (size_t)row * N + col;
                float v = acc[mt][nt][r] + bv;
                if (outF) outF[o] = v + (residual ? residual[o] : 0.0f);
                else      outH[o] = (_Float16)v;
            }
        }
    }
}

// ---------------------------------------------------------------------------
// Flash attention: one block per (head, 64-row q block). 128 threads (4 waves),
// each wave owns 16 q rows. Q and K fragments load straight from global (both
// are [row][d]-contiguous, matching the A/B fragment layouts); P is re-laid
// out through LDS to A-form; V is LDS-transposed to B-form [d][k'].
// ---------------------------------------------------------------------------
constexpr int PSTR = 72;  // LDS stride (halves) for P rows (144B, 16B-aligned)
constexpr int VSTR = 72;  // LDS stride (halves) for V^T rows

__global__ __launch_bounds__(128) void flash_attn_kernel(
    const _Float16* __restrict__ Qh, const _Float16* __restrict__ Kh,
    const _Float16* __restrict__ Vh, const float* __restrict__ mask,
    _Float16* __restrict__ Ctx, float scale)
{
    constexpr int S = 1024, H = 1024, HS = 64;
    __shared__ __attribute__((aligned(16))) _Float16 Plds[64 * PSTR];
    __shared__ __attribute__((aligned(16))) _Float16 Vt[64 * VSTR];

    const int qb   = blockIdx.x;          // 0..15  (64-row q block)
    const int head = blockIdx.y;          // 0..63  (b*NH + h)
    const int b = head >> 4, h = head & 15;
    const int tid  = threadIdx.x;
    const int wave = tid >> 5, lane = tid & 31;
    const int hf = lane >> 4, l16 = lane & 15;
    const int q0 = qb * 64;

    // Q fragments (A-layout): row = l16, K chunks split by lane half
    Frag aq[2];
    {
        const int srow = q0 + wave * 16 + l16;
        const _Float16* qp = Qh + ((size_t)(b * S + srow)) * H + h * HS;
        #pragma unroll
        for (int f = 0; f < 2; ++f) {
            aq[f].h8[0] = *(const half8*)(qp + f * 32 + hf * 8);
            aq[f].h8[1] = *(const half8*)(qp + f * 32 + 16 + hf * 8);
        }
    }

    f32x8 o[4] = {};
    float mrun[8], lrun[8];
    #pragma unroll
    for (int r = 0; r < 8; ++r) { mrun[r] = -3.0e38f; lrun[r] = 0.0f; }

    for (int kb = 0; kb < S / 64; ++kb) {
        const int kbase = kb * 64;
        // ---- S tile: scores[q, k'] = sum_d Q[q,d] K[k',d] ----
        f32x8 s[4] = {};
        #pragma unroll
        for (int nt = 0; nt < 4; ++nt) {
            const int kcol = kbase + nt * 16 + l16;
            const _Float16* kp = Kh + ((size_t)(b * S + kcol)) * H + h * HS + hf * 8;
            #pragma unroll
            for (int f = 0; f < 2; ++f) {
                Frag bk;
                bk.h8[0] = *(const half8*)(kp + f * 32);
                bk.h8[1] = *(const half8*)(kp + f * 32 + 16);
                s[nt] = wmma16x16x32(aq[f].v, bk.v, s[nt]);
            }
        }
        // ---- scale + attention mask ----
        #pragma unroll
        for (int nt = 0; nt < 4; ++nt) {
            const float mk = mask[b * S + kbase + nt * 16 + l16];
            #pragma unroll
            for (int r = 0; r < 8; ++r) s[nt][r] = s[nt][r] * scale + mk;
        }
        // ---- online softmax (row = r + 8*hf; its 16 cols live in one
        //      16-lane half, so xor-shuffles 1/2/4/8 reduce exactly that row) --
        #pragma unroll
        for (int r = 0; r < 8; ++r) {
            float mv = fmaxf(fmaxf(s[0][r], s[1][r]), fmaxf(s[2][r], s[3][r]));
            #pragma unroll
            for (int off = 8; off > 0; off >>= 1)
                mv = fmaxf(mv, __shfl_xor(mv, off, 32));
            const float mnew  = fmaxf(mrun[r], mv);
            const float alpha = __expf(mrun[r] - mnew);
            float rs = 0.0f;
            #pragma unroll
            for (int nt = 0; nt < 4; ++nt) {
                const float p = __expf(s[nt][r] - mnew);
                s[nt][r] = p;
                rs += p;
            }
            #pragma unroll
            for (int off = 8; off > 0; off >>= 1)
                rs += __shfl_xor(rs, off, 32);
            lrun[r] = lrun[r] * alpha + rs;
            mrun[r] = mnew;
            #pragma unroll
            for (int nt = 0; nt < 4; ++nt) o[nt][r] *= alpha;
        }

        __syncthreads();  // previous iteration's Plds/Vt reads complete
        // ---- P (C-layout regs) -> LDS f16 [m][k'] ----
        #pragma unroll
        for (int nt = 0; nt < 4; ++nt)
            #pragma unroll
            for (int r = 0; r < 8; ++r)
                Plds[(wave * 16 + r + hf * 8) * PSTR + nt * 16 + l16] =
                    (_Float16)s[nt][r];
        // ---- V block transpose -> Vt[d][k'] (B-layout source) ----
        {
            const int kr = tid >> 1, db = (tid & 1) * 32;
            const _Float16* vp =
                Vh + ((size_t)(b * S + kbase + kr)) * H + h * HS + db;
            #pragma unroll
            for (int i = 0; i < 32; i += 8) {
                half8 vv = *(const half8*)(vp + i);
                #pragma unroll
                for (int j = 0; j < 8; ++j)
                    Vt[(db + i + j) * VSTR + kr] = vv[j];
            }
        }
        __syncthreads();

        // ---- O += P @ V ----
        #pragma unroll
        for (int f = 0; f < 2; ++f) {
            Frag ap;
            const _Float16* pp = Plds + (wave * 16 + l16) * PSTR + f * 32;
            ap.h8[0] = *(const half8*)(pp + hf * 8);
            ap.h8[1] = *(const half8*)(pp + 16 + hf * 8);
            #pragma unroll
            for (int nt = 0; nt < 4; ++nt) {
                Frag bv;
                const _Float16* vv = Vt + (nt * 16 + l16) * VSTR + f * 32 + hf * 8;
                bv.h8[0] = *(const half8*)(vv);
                bv.h8[1] = *(const half8*)(vv + 16);
                o[nt] = wmma16x16x32(ap.v, bv.v, o[nt]);
            }
        }
    }

    // ---- normalize and store ctx (f16) ----
    #pragma unroll
    for (int nt = 0; nt < 4; ++nt)
        #pragma unroll
        for (int r = 0; r < 8; ++r) {
            const int srow = q0 + wave * 16 + r + hf * 8;
            const int d = nt * 16 + l16;
            Ctx[((size_t)(b * S + srow)) * H + h * HS + d] =
                (_Float16)(o[nt][r] / lrun[r]);
        }
}

// ---------------------------------------------------------------------------
// Row LayerNorm over 1024 elements (fp32), one 256-thread block per row.
// ---------------------------------------------------------------------------
__global__ __launch_bounds__(256) void layernorm_kernel(
    const float* __restrict__ Y, const float* __restrict__ gamma,
    const float* __restrict__ beta, float* __restrict__ out)
{
    const int row = blockIdx.x;
    const int tid = threadIdx.x;
    const float* y = Y + (size_t)row * 1024;
    f32x4 v = *(const f32x4*)(y + tid * 4);
    float s  = v[0] + v[1] + v[2] + v[3];
    float ss = v[0] * v[0] + v[1] * v[1] + v[2] * v[2] + v[3] * v[3];
    #pragma unroll
    for (int off = 16; off > 0; off >>= 1) {
        s  += __shfl_xor(s, off, 32);
        ss += __shfl_xor(ss, off, 32);
    }
    __shared__ float rs[8], rss[8];
    const int wave = tid >> 5, lane = tid & 31;
    if (lane == 0) { rs[wave] = s; rss[wave] = ss; }
    __syncthreads();
    s = 0.0f; ss = 0.0f;
    #pragma unroll
    for (int w = 0; w < 8; ++w) { s += rs[w]; ss += rss[w]; }
    const float mean = s * (1.0f / 1024.0f);
    const float var  = ss * (1.0f / 1024.0f) - mean * mean;
    const float rstd = rsqrtf(var + 1e-12f);
    f32x4 g  = *(const f32x4*)(gamma + tid * 4);
    f32x4 bt = *(const f32x4*)(beta + tid * 4);
    f32x4 ov;
    #pragma unroll
    for (int j = 0; j < 4; ++j) ov[j] = (v[j] - mean) * rstd * g[j] + bt[j];
    *(f32x4*)(out + (size_t)row * 1024 + tid * 4) = ov;
}

// ---------------------------------------------------------------------------
extern "C" void kernel_launch(void* const* d_in, const int* in_sizes, int n_in,
                              void* d_out, int out_size, void* d_ws, size_t ws_size,
                              hipStream_t stream)
{
    const float* hs    = (const float*)d_in[0];
    const float* mask  = (const float*)d_in[1];
    const float* qw    = (const float*)d_in[2];
    const float* qb    = (const float*)d_in[3];
    const float* kw    = (const float*)d_in[4];
    const float* kb    = (const float*)d_in[5];
    const float* vw    = (const float*)d_in[6];
    const float* vb    = (const float*)d_in[7];
    const float* ow    = (const float*)d_in[8];
    const float* ob    = (const float*)d_in[9];
    const float* gamma = (const float*)d_in[10];
    const float* beta  = (const float*)d_in[11];
    float* out = (float*)d_out;

    const int M = 4096, N = 1024, K = 1024;  // B*S, H, H
    const size_t MN = (size_t)M * N;
    const size_t NK = (size_t)N * K;

    _Float16* Xh  = (_Float16*)d_ws;     // f16 hidden states
    _Float16* Wqh = Xh + MN;             // f16 weights
    _Float16* Wkh = Wqh + NK;
    _Float16* Wvh = Wkh + NK;
    _Float16* Woh = Wvh + NK;
    _Float16* Qh  = Woh + NK;
    _Float16* Kh  = Qh + MN;
    _Float16* Vh  = Kh + MN;
    _Float16* Ctx = Vh + MN;
    float*    Y   = (float*)(Ctx + MN);

    // one-time fp32 -> fp16 conversions
    cvt_f32_f16_kernel<<<(int)(MN / 8 / 256), 256, 0, stream>>>(hs, Xh, (int)MN);
    cvt_f32_f16_kernel<<<(int)(NK / 8 / 256), 256, 0, stream>>>(qw, Wqh, (int)NK);
    cvt_f32_f16_kernel<<<(int)(NK / 8 / 256), 256, 0, stream>>>(kw, Wkh, (int)NK);
    cvt_f32_f16_kernel<<<(int)(NK / 8 / 256), 256, 0, stream>>>(vw, Wvh, (int)NK);
    cvt_f32_f16_kernel<<<(int)(NK / 8 / 256), 256, 0, stream>>>(ow, Woh, (int)NK);

    dim3 ggrid(N / GN, M / GM);  // (16, 32)
    wmma_gemm_async_kernel<<<ggrid, 256, 0, stream>>>(
        Xh, Wqh, qb, nullptr, Qh, nullptr, M, N, K);
    wmma_gemm_async_kernel<<<ggrid, 256, 0, stream>>>(
        Xh, Wkh, kb, nullptr, Kh, nullptr, M, N, K);
    wmma_gemm_async_kernel<<<ggrid, 256, 0, stream>>>(
        Xh, Wvh, vb, nullptr, Vh, nullptr, M, N, K);

    flash_attn_kernel<<<dim3(16, 64), 128, 0, stream>>>(
        Qh, Kh, Vh, mask, Ctx, 0.125f);

    wmma_gemm_async_kernel<<<ggrid, 256, 0, stream>>>(
        Ctx, Woh, ob, hs, nullptr, Y, M, N, K);

    layernorm_kernel<<<4096, 256, 0, stream>>>(Y, gamma, beta, out);
}
